// MTCNN_73203422593621
// MI455X (gfx1250) — matrix operand order repaired and checked by reading, not compiled, not run on hardware
//
#include <hip/hip_runtime.h>
#include <hip/hip_fp16.h>
#include <cmath>
#include <algorithm>

typedef _Float16 f16;
typedef __attribute__((ext_vector_type(16))) _Float16 v16h;
typedef __attribute__((ext_vector_type(8)))  _Float16 v8h;
typedef __attribute__((ext_vector_type(8)))  float    v8f;

#define CDIV(a,b) (((a)+(b)-1)/(b))
#define NEG_BIG (-3.0e38f)

// ---------------------------------------------------------------------------
// Area resize (matches jax area_matrix/area_resize) + normalize, NHWC f16, Cp=16
// ---------------------------------------------------------------------------
__global__ void k_area_resize(const float* __restrict__ img, int H, int W,
                              f16* __restrict__ out, int OH, int OW) {
  int t = blockIdx.x * blockDim.x + threadIdx.x;
  if (t >= OH * OW) return;
  int oy = t / OW, ox = t % OW;
  int sy = (int)(((long long)oy * H) / OH);
  int ey = (int)((((long long)(oy + 1)) * H + OH - 1) / OH);
  int sx = (int)(((long long)ox * W) / OW);
  int ex = (int)((((long long)(ox + 1)) * W + OW - 1) / OW);
  float inv = 1.0f / (float)((ey - sy) * (ex - sx));
  float acc0 = 0.f, acc1 = 0.f, acc2 = 0.f;
  for (int y = sy; y < ey; ++y)
    for (int x = sx; x < ex; ++x) {
      const float* p = img + ((size_t)y * W + x) * 3;
      acc0 += p[0]; acc1 += p[1]; acc2 += p[2];
    }
  f16* o = out + (size_t)t * 16;
  o[0] = (f16)((acc0 * inv - 127.5f) * 0.0078125f);
  o[1] = (f16)((acc1 * inv - 127.5f) * 0.0078125f);
  o[2] = (f16)((acc2 * inv - 127.5f) * 0.0078125f);
  for (int c = 3; c < 16; ++c) o[c] = (f16)0.f;
}

// ---------------------------------------------------------------------------
// Pack conv/dense weights OIHW f32 -> f16 in WMMA B-FRAGMENT order:
//   frag[(nt*nKc + kc)*512 + lane*16 + e]
//   with K = kc*32 + e + (lane>=16 ? 16 : 0), co = nt*16 + (lane&15),
//   K decomposed as (ky*kw + kx)*Cpad + ci.
// Each lane's 32-byte B fragment is then contiguous in memory.
// ---------------------------------------------------------------------------
__global__ void k_pack_frag(const float* __restrict__ w, int Cout, int Cin,
                            int kh, int kw, int Cpad, int Kpad, int CoPad,
                            f16* __restrict__ frag) {
  int nKc = Kpad >> 5;
  int total = (CoPad >> 4) * nKc * 512;
  int t = blockIdx.x * blockDim.x + threadIdx.x;
  if (t >= total) return;
  int e = t & 15;
  int lane = (t >> 4) & 31;
  int chunk = t >> 9;                 // nt*nKc + kc
  int nt = chunk / nKc, kc = chunk - nt * nKc;
  int K = kc * 32 + e + ((lane >= 16) ? 16 : 0);
  int co = nt * 16 + (lane & 15);
  float v = 0.f;
  if (K < kh * kw * Cpad && co < Cout) {
    int ky = K / (kw * Cpad);
    int r  = K - ky * kw * Cpad;
    int kx = r / Cpad;
    int ci = r - kx * Cpad;
    if (ci < Cin) v = w[(((size_t)co * Cin + ci) * kh + ky) * kw + kx];
  }
  frag[t] = (f16)v;
}

// ---------------------------------------------------------------------------
// Implicit-GEMM conv via v_wmma_f32_16x16x32_f16.
//  - A: activations NHWC f16 (channel-padded, padded chans zero); each lane's
//    fragment = two aligned 16B global_load_b128 (K-groups of 8 stay inside
//    one (ky,kx) cell because Cpad % 16 == 0 and group bases are 8-aligned).
//  - B: weights pre-packed in fragment order. All 8 waves of a block share one
//    N-tile (blockIdx.y); wave 0 streams the next K-chunk's 1KB fragment into
//    double-buffered LDS via global_load_async_to_lds_b128 (ASYNCcnt), waits
//    with s_wait_asynccnt, and __syncthreads publishes it; every wave then
//    reads its 32B with ds_load_b128.
//  - Fused bias + PReLU; writes f16 (next-layer act) and/or f32 (heads).
// ---------------------------------------------------------------------------
__global__ void __launch_bounds__(256)
k_wmma_conv(const f16* __restrict__ A, int N, int H, int W, int Cp,
            const f16* __restrict__ Bf, int kh, int kw, int Kpad,
            int CoPad, int Cout,
            const float* __restrict__ bias, const float* __restrict__ pr,
            f16* __restrict__ outH, float* __restrict__ outF,
            int OH, int OW) {
  __shared__ __align__(16) f16 ldsB[2][512];   // 2 x 1KB double buffer

  int wave = threadIdx.x >> 5;
  int lane = threadIdx.x & 31;
  int laneHi = lane >> 4;                      // 0: lanes 0-15, 1: lanes 16-31
  int ln = lane & 15;
  int nt = blockIdx.y;                          // shared N-tile for the block
  int Mtot = N * OH * OW;
  int mt = blockIdx.x * 8 + wave;               // M-tile of this wave
  int m_row = mt * 16 + ln;
  int valid_m = (m_row < Mtot);
  int bb = 0, oy = 0, ox = 0;
  if (valid_m) {
    bb = m_row / (OH * OW);
    int r = m_row - bb * OH * OW;
    oy = r / OW; ox = r - oy * OW;
  }
  int KV = kh * kw * Cp;
  int nKc = Kpad >> 5;
  const f16* Bbase = Bf + (size_t)nt * nKc * 512;

  // stage chunk 0 (wave 0 only; 32 lanes x 32B = 1KB via 2 async b128 each)
  if (wave == 0) {
    unsigned long long ga = (unsigned long long)(Bbase + (size_t)lane * 16);
    unsigned lo = (unsigned)(size_t)&ldsB[0][lane * 16];
    asm volatile("global_load_async_to_lds_b128 %0, %1, off"
                 :: "v"(lo), "v"(ga) : "memory");
    asm volatile("global_load_async_to_lds_b128 %0, %1, off"
                 :: "v"(lo + 16u), "v"(ga + 16ull) : "memory");
  }

  v8f acc;
#pragma unroll
  for (int i = 0; i < 8; ++i) acc[i] = 0.f;

  for (int kc = 0; kc < nKc; ++kc) {
    if (wave == 0)
      asm volatile("s_wait_asynccnt 0x0" ::: "memory");
    __syncthreads();                           // publish buffer kc&1
    if (wave == 0 && kc + 1 < nKc) {           // prefetch next chunk
      unsigned long long ga =
          (unsigned long long)(Bbase + ((size_t)(kc + 1) * 512 + lane * 16));
      unsigned lo = (unsigned)(size_t)&ldsB[(kc + 1) & 1][lane * 16];
      asm volatile("global_load_async_to_lds_b128 %0, %1, off"
                   :: "v"(lo), "v"(ga) : "memory");
      asm volatile("global_load_async_to_lds_b128 %0, %1, off"
                   :: "v"(lo + 16u), "v"(ga + 16ull) : "memory");
    }

    // ---- B fragment from LDS: 2 x ds_load_b128
    const v8h* lb = (const v8h*)&ldsB[kc & 1][lane * 16];
    v8h blo = lb[0];
    v8h bhi = lb[1];
    v16h bfr = __builtin_shufflevector(blo, bhi,
        0, 1, 2, 3, 4, 5, 6, 7, 8, 9, 10, 11, 12, 13, 14, 15);

    // ---- A fragment: 2 aligned 16B loads (groups of 8 consecutive K)
    int k0 = kc * 32;
    int base1 = k0 + laneHi * 16;       // elements 0..7
    int base2 = k0 + laneHi * 16 + 8;   // elements 8..15
    v8h alo, ahi;
#pragma unroll
    for (int i = 0; i < 8; ++i) { alo[i] = (f16)0.f; ahi[i] = (f16)0.f; }
    if (valid_m && base1 < KV) {
      int ky = base1 / (kw * Cp);
      int r  = base1 - ky * kw * Cp;
      int kx = r / Cp;
      int ci = r - kx * Cp;
      alo = *(const v8h*)(A + ((size_t)((bb * H + oy + ky) * W) + (ox + kx)) * Cp + ci);
    }
    if (valid_m && base2 < KV) {
      int ky = base2 / (kw * Cp);
      int r  = base2 - ky * kw * Cp;
      int kx = r / Cp;
      int ci = r - kx * Cp;
      ahi = *(const v8h*)(A + ((size_t)((bb * H + oy + ky) * W) + (ox + kx)) * Cp + ci);
    }
    v16h afr = __builtin_shufflevector(alo, ahi,
        0, 1, 2, 3, 4, 5, 6, 7, 8, 9, 10, 11, 12, 13, 14, 15);

    acc = __builtin_amdgcn_wmma_f32_16x16x32_f16(false, afr, false, bfr,
                                                 (short)0, acc, false, false);
  }

  // ---- Epilogue: VGPR r -> M = r + (laneHi?8:0), N = lane&15
  int co = nt * 16 + ln;
#pragma unroll
  for (int r = 0; r < 8; ++r) {
    int m = mt * 16 + r + (laneHi ? 8 : 0);
    if (m >= Mtot) continue;
    float v = acc[r];
    if (co < Cout) {
      v += bias[co];
      if (pr) v = v > 0.f ? v : pr[co] * v;
    } else {
      v = 0.f;  // keep padded channels zero for the next layer
    }
    if (outH) outH[(size_t)m * CoPad + co] = (f16)v;
    if (outF) outF[(size_t)m * CoPad + co] = v;
  }
}

// ---------------------------------------------------------------------------
// Ceil-mode max pool (reference pads with -inf == clamp window)
// ---------------------------------------------------------------------------
__global__ void k_maxpool(const f16* __restrict__ in, int N, int H, int W, int Cp,
                          f16* __restrict__ out, int OH, int OW, int k, int s) {
  int t = blockIdx.x * blockDim.x + threadIdx.x;
  if (t >= N * OH * OW * Cp) return;
  int c = t % Cp; int r = t / Cp;
  int ox = r % OW; r /= OW;
  int oy = r % OH; int b = r / OH;
  float m = NEG_BIG;
  for (int dy = 0; dy < k; ++dy) {
    int y = oy * s + dy; if (y >= H) break;
    for (int dx = 0; dx < k; ++dx) {
      int x = ox * s + dx; if (x >= W) break;
      float v = (float)in[((size_t)((b * H + y) * W) + x) * Cp + c];
      m = fmaxf(m, v);
    }
  }
  out[(size_t)t] = (f16)m;
}

// ---------------------------------------------------------------------------
// 2-class softmax -> face score (softmax[1] == sigmoid(x1 - x0))
// ---------------------------------------------------------------------------
__global__ void k_pnet_score(const float* __restrict__ prob, int n,
                             float* __restrict__ score) {
  int t = blockIdx.x * blockDim.x + threadIdx.x;
  if (t >= n) return;
  float p0 = prob[(size_t)t * 16 + 0], p1 = prob[(size_t)t * 16 + 1];
  score[t] = 1.f / (1.f + expf(p0 - p1));
}

// ---------------------------------------------------------------------------
// Deterministic iterative-argmax top-K (ties -> smallest index, = lax.top_k).
// Single block of 256 threads; also produces score-descending order arrays.
// ---------------------------------------------------------------------------
__global__ void k_topk(const float* __restrict__ keys, int n, int K,
                       int* __restrict__ mask,
                       float* __restrict__ vals, int* __restrict__ idxs) {
  __shared__ float sv[256];
  __shared__ int   si[256];
  int tid = threadIdx.x;
  for (int i = tid; i < n; i += 256) mask[i] = 0;
  __syncthreads();
  for (int t = 0; t < K; ++t) {
    float bv = NEG_BIG; int bi = 0x7FFFFFFF;
    for (int i = tid; i < n; i += 256) {
      if (!mask[i]) {
        float v = keys[i];
        if (v > bv || (v == bv && i < bi)) { bv = v; bi = i; }
      }
    }
    sv[tid] = bv; si[tid] = bi;
    __syncthreads();
    for (int o = 128; o > 0; o >>= 1) {
      if (tid < o) {
        if (sv[tid + o] > sv[tid] ||
            (sv[tid + o] == sv[tid] && si[tid + o] < si[tid])) {
          sv[tid] = sv[tid + o]; si[tid] = si[tid + o];
        }
      }
      __syncthreads();
    }
    if (tid == 0) {
      if (vals) vals[t] = sv[0];
      if (idxs) idxs[t] = (si[0] == 0x7FFFFFFF) ? -1 : si[0];
      if (si[0] != 0x7FFFFFFF) mask[si[0]] = 1;
    }
    __syncthreads();
  }
}

// ---------------------------------------------------------------------------
// Build stage-1 candidate rows [x1,y1,x2,y2,score,reg0..3]
// ---------------------------------------------------------------------------
__global__ void k_build_b9(const float* __restrict__ vals, const int* __restrict__ idxs,
                           const float* __restrict__ regb, int n, int mw,
                           float scale, float thr, int K,
                           float* __restrict__ b9, int* __restrict__ valid) {
  int t = blockIdx.x * blockDim.x + threadIdx.x;
  if (t >= 384) return;
  float* row = b9 + (size_t)t * 9;
  int id = (t < K) ? idxs[t] : -1;
  if (id < 0 || id >= n) {
    for (int c = 0; c < 9; ++c) row[c] = 0.f;
    valid[t] = 0;
    return;
  }
  float v = vals[t];
  float yy = (float)(id / mw), xx = (float)(id % mw);
  row[0] = (2.f * xx + 1.f) / scale;
  row[1] = (2.f * yy + 1.f) / scale;
  row[2] = (2.f * xx + 12.f) / scale;
  row[3] = (2.f * yy + 12.f) / scale;
  row[4] = v;
  for (int c = 0; c < 4; ++c) row[5 + c] = regb[(size_t)id * 16 + c];
  valid[t] = (v >= thr) ? 1 : 0;
}

// ---------------------------------------------------------------------------
// NMS. Boxes processed in score-descending order (ord==NULL -> identity).
// mode 0 = union, mode 1 = min.  Single block.
// ---------------------------------------------------------------------------
__global__ void k_nms(const float* __restrict__ boxes, int stride, int n,
                      const int* __restrict__ ord, int* __restrict__ valid,
                      float thr, int mode) {
  __shared__ unsigned char keep[4608];
  int tid = threadIdx.x;
  int bd = blockDim.x;
  for (int i = tid; i < n; i += bd) {
    int gi = ord ? ord[i] : i;
    keep[i] = valid[gi] ? 1 : 0;
  }
  __syncthreads();
  float off = (mode == 1) ? 1.f : 0.f;
  for (int i = 0; i < n; ++i) {
    __syncthreads();
    if (!keep[i]) continue;                 // shared value: uniform branch
    int gi = ord ? ord[i] : i;
    float x1 = boxes[(size_t)gi * stride + 0];
    float y1 = boxes[(size_t)gi * stride + 1];
    float x2 = boxes[(size_t)gi * stride + 2];
    float y2 = boxes[(size_t)gi * stride + 3];
    float ai = (x2 - x1 + off) * (y2 - y1 + off);
    for (int j = i + 1 + tid; j < n; j += bd) {
      if (!keep[j]) continue;
      int gj = ord ? ord[j] : j;
      float bx1 = boxes[(size_t)gj * stride + 0];
      float by1 = boxes[(size_t)gj * stride + 1];
      float bx2 = boxes[(size_t)gj * stride + 2];
      float by2 = boxes[(size_t)gj * stride + 3];
      float aj = (bx2 - bx1 + off) * (by2 - by1 + off);
      float iw = fmaxf(fminf(x2, bx2) - fmaxf(x1, bx1) + off, 0.f);
      float ih = fmaxf(fminf(y2, by2) - fmaxf(y1, by1) + off, 0.f);
      float inter = iw * ih;
      float denom = (mode == 1) ? fminf(ai, aj) : (ai + aj - inter);
      float iou = inter / fmaxf(denom, 1e-9f);
      if (iou > thr) keep[j] = 0;
    }
  }
  __syncthreads();
  for (int i = tid; i < n; i += bd) {
    int gi = ord ? ord[i] : i;
    valid[gi] = keep[i];
  }
}

// keys[i] = valid ? score : -big   (score at column 4)
__global__ void k_keys(const float* __restrict__ boxes, int stride,
                       const int* __restrict__ valid, int n,
                       float* __restrict__ keys) {
  int t = blockIdx.x * blockDim.x + threadIdx.x;
  if (t >= n) return;
  keys[t] = valid[t] ? boxes[(size_t)t * stride + 4] : -1e30f;
}

// stage-1 regression + rerec
__global__ void k_refine1(const float* __restrict__ b9, int n,
                          float* __restrict__ out5) {
  int t = blockIdx.x * blockDim.x + threadIdx.x;
  if (t >= n) return;
  const float* r = b9 + (size_t)t * 9;
  float regw = r[2] - r[0], regh = r[3] - r[1];
  float x1 = r[0] + r[5] * regw, y1 = r[1] + r[6] * regh;
  float x2 = r[2] + r[7] * regw, y2 = r[3] + r[8] * regh;
  float h = y2 - y1, w = x2 - x1, l = fmaxf(h, w);
  float nx1 = x1 + w * 0.5f - l * 0.5f;
  float ny1 = y1 + h * 0.5f - l * 0.5f;
  float* o = out5 + (size_t)t * 5;
  o[0] = nx1; o[1] = ny1; o[2] = nx1 + l; o[3] = ny1 + l; o[4] = r[4];
}

__global__ void k_gather(const float* __restrict__ in5, const int* __restrict__ invalid_,
                         const int* __restrict__ idxs, int R,
                         float* __restrict__ out5, int* __restrict__ outvalid) {
  int t = blockIdx.x * blockDim.x + threadIdx.x;
  if (t >= R) return;
  int id = idxs[t]; if (id < 0) id = 0;
  for (int c = 0; c < 5; ++c) out5[(size_t)t * 5 + c] = in5[(size_t)id * 5 + c];
  outvalid[t] = (idxs[t] >= 0) ? invalid_[id] : 0;
}

// ---------------------------------------------------------------------------
// pad_boxes + bilinear crop_resize + normalize, NHWC f16 Cp=16
// ---------------------------------------------------------------------------
__global__ void k_crop(const float* __restrict__ img, int H, int W,
                       const float* __restrict__ boxes, int R, int out,
                       f16* __restrict__ dst, int* __restrict__ okc_out) {
  int t = blockIdx.x * blockDim.x + threadIdx.x;
  if (t >= R * out * out) return;
  int j = t % out; int rr = t / out;
  int i = rr % out; int r = rr / out;
  const float* b = boxes + (size_t)r * 5;
  float b0 = truncf(b[0]), b1 = truncf(b[1]), b2 = truncf(b[2]), b3 = truncf(b[3]);
  float x  = fmaxf(b0, 1.f), y = fmaxf(b1, 1.f);
  float ex = fminf(b2, (float)W), ey = fminf(b3, (float)H);
  int ok = (ey > y - 1.f) && (ex > x - 1.f);
  float ht = fmaxf(ey - y + 1.f, 1.f), wd = fmaxf(ex - x + 1.f, 1.f);
  float y0 = y - 1.f, x0 = x - 1.f;
  float ys = y0 + ((float)i + 0.5f) * ht / (float)out - 0.5f;
  float xs = x0 + ((float)j + 0.5f) * wd / (float)out - 0.5f;
  float yf = floorf(ys), xf = floorf(xs);
  float wy = ys - yf, wx = xs - xf;
  int y0c = min(max((int)yf, 0), H - 1);
  int y1c = min(y0c + 1, H - 1);
  int x0c = min(max((int)xf, 0), W - 1);
  int x1c = min(x0c + 1, W - 1);
  f16* o = dst + (size_t)t * 16;
  float okf = ok ? 1.f : 0.f;
  for (int c = 0; c < 3; ++c) {
    float Ia = img[((size_t)y0c * W + x0c) * 3 + c];
    float Ib = img[((size_t)y0c * W + x1c) * 3 + c];
    float Ic = img[((size_t)y1c * W + x0c) * 3 + c];
    float Id = img[((size_t)y1c * W + x1c) * 3 + c];
    float top = Ia * (1.f - wx) + Ib * wx;
    float bot = Ic * (1.f - wx) + Id * wx;
    float v = top * (1.f - wy) + bot * wy;
    o[c] = (f16)((v - 127.5f) * 0.0078125f * okf);
  }
  for (int c = 3; c < 16; ++c) o[c] = (f16)0.f;
  if (i == 0 && j == 0) okc_out[r] = ok;
}

// _flat_t: NHWC f16 -> [N][K] f16 with k = (x*H + y)*C + c
__global__ void k_flatten(const f16* __restrict__ in, int N, int H, int W,
                          int Cp, int C, f16* __restrict__ out) {
  int K = W * H * C;
  int t = blockIdx.x * blockDim.x + threadIdx.x;
  if (t >= N * K) return;
  int k = t % K; int n = t / K;
  int x = k / (H * C);
  int rem = k - x * H * C;
  int y = rem / C;
  int c = rem - y * C;
  out[(size_t)n * K + k] = in[((size_t)((n * H + y) * W) + x) * Cp + c];
}

__global__ void k_rnet_filter(const float* __restrict__ prob, const int* __restrict__ okc,
                              int R, float thr, float* __restrict__ boxes,
                              int* __restrict__ valid) {
  int t = blockIdx.x * blockDim.x + threadIdx.x;
  if (t >= R) return;
  float p0 = prob[(size_t)t * 16 + 0], p1 = prob[(size_t)t * 16 + 1];
  float s = 1.f / (1.f + expf(p0 - p1));
  boxes[(size_t)t * 5 + 4] = s;
  valid[t] = (valid[t] && okc[t] && (s > thr)) ? 1 : 0;
}

// stage-2 bbreg + rerec (in place)
__global__ void k_bbreg_rerec(float* __restrict__ boxes, const float* __restrict__ regb,
                              int n) {
  int t = blockIdx.x * blockDim.x + threadIdx.x;
  if (t >= n) return;
  float* b = boxes + (size_t)t * 5;
  const float* rg = regb + (size_t)t * 16;
  float w = b[2] - b[0] + 1.f, h = b[3] - b[1] + 1.f;
  float x1 = b[0] + rg[0] * w, y1 = b[1] + rg[1] * h;
  float x2 = b[2] + rg[2] * w, y2 = b[3] + rg[3] * h;
  float hh = y2 - y1, ww = x2 - x1, l = fmaxf(hh, ww);
  float nx1 = x1 + ww * 0.5f - l * 0.5f;
  float ny1 = y1 + hh * 0.5f - l * 0.5f;
  b[0] = nx1; b[1] = ny1; b[2] = nx1 + l; b[3] = ny1 + l;
}

// O-Net final: score3, landmark points, bbreg (no rerec)
__global__ void k_onet_final(const float* __restrict__ prob, const float* __restrict__ regb,
                             const float* __restrict__ pts, const int* __restrict__ okc,
                             float thr, const float* __restrict__ obox,
                             int* __restrict__ ovalid, float* __restrict__ fbox,
                             float* __restrict__ fpts) {
  int t = blockIdx.x * blockDim.x + threadIdx.x;
  if (t >= 128) return;
  float p0 = prob[(size_t)t * 16 + 0], p1 = prob[(size_t)t * 16 + 1];
  float s = 1.f / (1.f + expf(p0 - p1));
  ovalid[t] = (ovalid[t] && okc[t] && (s > thr)) ? 1 : 0;
  const float* b = obox + (size_t)t * 5;
  float wi = b[2] - b[0] + 1.f, hi = b[3] - b[1] + 1.f;
  for (int k = 0; k < 5; ++k) {
    fpts[(size_t)t * 10 + k]     = wi * pts[(size_t)t * 16 + k] + b[0] - 1.f;
    fpts[(size_t)t * 10 + 5 + k] = hi * pts[(size_t)t * 16 + 5 + k] + b[1] - 1.f;
  }
  const float* rg = regb + (size_t)t * 16;
  float* fb = fbox + (size_t)t * 5;
  fb[0] = b[0] + rg[0] * wi;
  fb[1] = b[1] + rg[1] * hi;
  fb[2] = b[2] + rg[2] * wi;
  fb[3] = b[3] + rg[3] * hi;
  fb[4] = s;
}

__global__ void k_output(const float* __restrict__ fbox, const float* __restrict__ fpts,
                         const int* __restrict__ valid, float* __restrict__ out) {
  int t = blockIdx.x * blockDim.x + threadIdx.x;
  if (t >= 128) return;
  float vf = valid[t] ? 1.f : 0.f;
  for (int c = 0; c < 5; ++c) out[t * 5 + c] = fbox[(size_t)t * 5 + c] * vf;
  for (int k = 0; k < 5; ++k) {
    out[640 + t * 10 + k * 2 + 0] = fpts[(size_t)t * 10 + k] * vf;      // px
    out[640 + t * 10 + k * 2 + 1] = fpts[(size_t)t * 10 + 5 + k] * vf;  // py
  }
}

// ---------------------------------------------------------------------------
// Host orchestration
// ---------------------------------------------------------------------------
static inline int pooled_dim(int x, int k, int s) {
  int v = x - k;
  return (v + s - 1) / s + 1;   // ceil((x-k)/s)+1, v >= 0 in this pipeline
}

extern "C" void kernel_launch(void* const* d_in, const int* in_sizes, int n_in,
                              void* d_out, int out_size, void* d_ws, size_t ws_size,
                              hipStream_t stream) {
  (void)in_sizes; (void)n_in; (void)out_size; (void)ws_size;
  const int IMG_H = 1080, IMG_W = 1920;
  const float* img = (const float*)d_in[0];

  struct L { const float *a, *b, *w; };
  auto layer = [&](int base) -> L {
    return L{ (const float*)d_in[base], (const float*)d_in[base + 1],
              (const float*)d_in[base + 2] };
  };
  // pytree (alphabetical) leaf order: onet, pnet, rnet; per-layer a, b, w
  L o_c1 = layer(1),  o_c2 = layer(4),  o_c3 = layer(7),  o_c4 = layer(10);
  L o_d5 = layer(13), o_d61 = layer(16), o_d62 = layer(19), o_d63 = layer(22);
  L p_c1 = layer(25), p_c2 = layer(28), p_c3 = layer(31);
  L p_c41 = layer(34), p_c42 = layer(37);
  L r_c1 = layer(40), r_c2 = layer(43), r_c3 = layer(46);
  L r_d4 = layer(49), r_d51 = layer(52), r_d52 = layer(55);

  // ---- fixed deterministic bump allocator over d_ws
  char* wsb = (char*)d_ws;
  size_t cur = 0;
  auto alloc = [&](size_t bytes) -> char* {
    size_t o = (cur + 255) & ~(size_t)255;
    cur = o + bytes;
    return wsb + o;
  };
  const size_t ACT_BYTES = (size_t)26 << 20;   // covers 649*1153*16 f16
  f16*   ACT_A = (f16*)alloc(ACT_BYTES);
  f16*   ACT_B = (f16*)alloc(ACT_BYTES);
  f16*   ACT_C = (f16*)alloc(ACT_BYTES);
  float* HPROB = (float*)alloc((size_t)12 << 20);  // [pos][16] f32
  float* HREG  = (float*)alloc((size_t)12 << 20);
  float* HPTS  = (float*)alloc(128 * 16 * 4);
  float* SCORE = (float*)alloc((size_t)1 << 20);
  int*   MASK  = (int*)alloc((size_t)1 << 20);
  float* VALS  = (float*)alloc(2048);
  int*   IDXS  = (int*)alloc(2048);
  float* B9    = (float*)alloc(4608 * 9 * 4);
  int*   VALID = (int*)alloc(4608 * 4);
  int*   ORD   = (int*)alloc(4608 * 4);
  float* KEYS  = (float*)alloc(4608 * 4);
  float* BOX5  = (float*)alloc(4608 * 5 * 4);
  float* RBOX  = (float*)alloc(256 * 5 * 4);
  int*   RVALID = (int*)alloc(256 * 4);
  float* OBOX  = (float*)alloc(128 * 5 * 4);
  int*   OVALID = (int*)alloc(128 * 4);
  int*   OKC   = (int*)alloc(256 * 4);
  float* FBOX  = (float*)alloc(128 * 5 * 4);
  float* FPTS  = (float*)alloc(128 * 10 * 4);

  // ---- pack all weights into WMMA B-fragment order -- stateless, every call
  auto pack = [&](const float* w, int Cout, int Cin, int kh, int kw,
                  int Cpad, int Kpad, int CoPad) -> f16* {
    f16* wp = (f16*)alloc((size_t)Kpad * CoPad * sizeof(f16));
    int n = Kpad * CoPad;    // == (CoPad/16)*(Kpad/32)*512
    k_pack_frag<<<CDIV(n, 256), 256, 0, stream>>>(w, Cout, Cin, kh, kw,
                                                  Cpad, Kpad, CoPad, wp);
    return wp;
  };
  f16* Wpc1  = pack(p_c1.w, 10, 3, 3, 3, 16, 160, 16);
  f16* Wpc2  = pack(p_c2.w, 16, 10, 3, 3, 16, 160, 16);
  f16* Wpc3  = pack(p_c3.w, 32, 16, 3, 3, 16, 160, 32);
  f16* Wpc41 = pack(p_c41.w, 2, 32, 1, 1, 32, 32, 16);
  f16* Wpc42 = pack(p_c42.w, 4, 32, 1, 1, 32, 32, 16);
  f16* Wrc1  = pack(r_c1.w, 28, 3, 3, 3, 16, 160, 32);
  f16* Wrc2  = pack(r_c2.w, 48, 28, 3, 3, 32, 288, 48);
  f16* Wrc3  = pack(r_c3.w, 64, 48, 2, 2, 48, 192, 64);
  f16* Wrd4  = pack(r_d4.w, 128, 576, 1, 1, 576, 576, 128);
  f16* Wrd51 = pack(r_d51.w, 2, 128, 1, 1, 128, 128, 16);
  f16* Wrd52 = pack(r_d52.w, 4, 128, 1, 1, 128, 128, 16);
  f16* Woc1  = pack(o_c1.w, 32, 3, 3, 3, 16, 160, 32);
  f16* Woc2  = pack(o_c2.w, 64, 32, 3, 3, 32, 288, 64);
  f16* Woc3  = pack(o_c3.w, 64, 64, 3, 3, 64, 576, 64);
  f16* Woc4  = pack(o_c4.w, 128, 64, 2, 2, 64, 256, 128);
  f16* Wod5  = pack(o_d5.w, 256, 1152, 1, 1, 1152, 1152, 256);
  f16* Wod61 = pack(o_d61.w, 2, 256, 1, 1, 256, 256, 16);
  f16* Wod62 = pack(o_d62.w, 4, 256, 1, 1, 256, 256, 16);
  f16* Wod63 = pack(o_d63.w, 10, 256, 1, 1, 256, 256, 16);

  auto conv = [&](const f16* Ain, int N, int H, int W, int Cp, const f16* Wp,
                  int kh, int kw, int Kpad, int CoPad, int Cout,
                  const float* bias, const float* pr, f16* outH, float* outF,
                  int OH, int OW) {
    int Mtot = N * OH * OW;
    dim3 grid(CDIV(CDIV(Mtot, 16), 8), CoPad / 16);
    k_wmma_conv<<<grid, 256, 0, stream>>>(Ain, N, H, W, Cp, Wp, kh, kw, Kpad,
                                          CoPad, Cout, bias, pr, outH, outF,
                                          OH, OW);
  };
  auto pool = [&](const f16* in, int N, int H, int W, int Cp, f16* out,
                  int OH, int OW, int k, int s) {
    int n = N * OH * OW * Cp;
    k_maxpool<<<CDIV(n, 256), 256, 0, stream>>>(in, N, H, W, Cp, out, OH, OW, k, s);
  };

  // ================= Stage 1: P-Net over image pyramid =================
  const double m = 12.0 / 20.0;
  const double factor = 0.709;
  double minl = (double)std::min(IMG_H, IMG_W) * m;
  int n_scales = (int)(std::log(12.0 / minl) / std::log(factor)) + 1;  // 12

  for (int si = 0; si < n_scales; ++si) {
    double sc = m * std::pow(factor, (double)si);
    int oh = (int)(IMG_H * sc + 1.0), ow = (int)(IMG_W * sc + 1.0);
    int h1 = oh - 2, w1 = ow - 2;
    int hp = pooled_dim(h1, 2, 2), wp = pooled_dim(w1, 2, 2);
    int h2 = hp - 2, w2 = wp - 2;
    int h3 = h2 - 2, w3 = w2 - 2;

    k_area_resize<<<CDIV(oh * ow, 256), 256, 0, stream>>>(img, IMG_H, IMG_W,
                                                          ACT_A, oh, ow);
    conv(ACT_A, 1, oh, ow, 16, Wpc1, 3, 3, 160, 16, 10, p_c1.b, p_c1.a,
         ACT_B, nullptr, h1, w1);
    pool(ACT_B, 1, h1, w1, 16, ACT_C, hp, wp, 2, 2);
    conv(ACT_C, 1, hp, wp, 16, Wpc2, 3, 3, 160, 16, 16, p_c2.b, p_c2.a,
         ACT_A, nullptr, h2, w2);
    conv(ACT_A, 1, h2, w2, 16, Wpc3, 3, 3, 160, 32, 32, p_c3.b, p_c3.a,
         ACT_B, nullptr, h3, w3);
    conv(ACT_B, 1, h3, w3, 32, Wpc41, 1, 1, 32, 16, 2, p_c41.b, nullptr,
         nullptr, HPROB, h3, w3);
    conv(ACT_B, 1, h3, w3, 32, Wpc42, 1, 1, 32, 16, 4, p_c42.b, nullptr,
         nullptr, HREG, h3, w3);

    int n = h3 * w3;
    k_pnet_score<<<CDIV(n, 256), 256, 0, stream>>>(HPROB, n, SCORE);
    int K = std::min(384, n);
    k_topk<<<1, 256, 0, stream>>>(SCORE, n, K, MASK, VALS, IDXS);
    k_build_b9<<<CDIV(384, 256), 256, 0, stream>>>(VALS, IDXS, HREG, n, w3,
                                                   (float)sc, 0.6f, K,
                                                   B9 + (size_t)si * 384 * 9,
                                                   VALID + si * 384);
    // per-scale NMS(0.5): topk output is already score-descending
    k_nms<<<1, 512, 0, stream>>>(B9 + (size_t)si * 384 * 9, 9, 384, nullptr,
                                 VALID + si * 384, 0.5f, 0);
  }

  int NB = n_scales * 384;   // 4608
  // cross-scale NMS(0.7) on score-descending order
  k_keys<<<CDIV(NB, 256), 256, 0, stream>>>(B9, 9, VALID, NB, KEYS);
  k_topk<<<1, 256, 0, stream>>>(KEYS, NB, NB, MASK, nullptr, ORD);
  k_nms<<<1, 512, 0, stream>>>(B9, 9, NB, ORD, VALID, 0.7f, 0);
  k_refine1<<<CDIV(NB, 256), 256, 0, stream>>>(B9, NB, BOX5);
  // take_top R_RNET = 256
  k_keys<<<CDIV(NB, 256), 256, 0, stream>>>(BOX5, 5, VALID, NB, KEYS);
  k_topk<<<1, 256, 0, stream>>>(KEYS, NB, 256, MASK, VALS, IDXS);
  k_gather<<<1, 256, 0, stream>>>(BOX5, VALID, IDXS, 256, RBOX, RVALID);

  // ================= Stage 2: R-Net on 256 x 24x24 crops =================
  k_crop<<<CDIV(256 * 24 * 24, 256), 256, 0, stream>>>(img, IMG_H, IMG_W,
                                                       RBOX, 256, 24, ACT_A, OKC);
  conv(ACT_A, 256, 24, 24, 16, Wrc1, 3, 3, 160, 32, 28, r_c1.b, r_c1.a,
       ACT_B, nullptr, 22, 22);
  pool(ACT_B, 256, 22, 22, 32, ACT_C, 11, 11, 3, 2);
  conv(ACT_C, 256, 11, 11, 32, Wrc2, 3, 3, 288, 48, 48, r_c2.b, r_c2.a,
       ACT_A, nullptr, 9, 9);
  pool(ACT_A, 256, 9, 9, 48, ACT_B, 4, 4, 3, 2);
  conv(ACT_B, 256, 4, 4, 48, Wrc3, 2, 2, 192, 64, 64, r_c3.b, r_c3.a,
       ACT_C, nullptr, 3, 3);
  k_flatten<<<CDIV(256 * 576, 256), 256, 0, stream>>>(ACT_C, 256, 3, 3, 64, 64, ACT_A);
  conv(ACT_A, 256, 1, 1, 576, Wrd4, 1, 1, 576, 128, 128, r_d4.b, r_d4.a,
       ACT_B, nullptr, 1, 1);
  conv(ACT_B, 256, 1, 1, 128, Wrd51, 1, 1, 128, 16, 2, r_d51.b, nullptr,
       nullptr, HPROB, 1, 1);
  conv(ACT_B, 256, 1, 1, 128, Wrd52, 1, 1, 128, 16, 4, r_d52.b, nullptr,
       nullptr, HREG, 1, 1);
  k_rnet_filter<<<1, 256, 0, stream>>>(HPROB, OKC, 256, 0.7f, RBOX, RVALID);
  k_keys<<<1, 256, 0, stream>>>(RBOX, 5, RVALID, 256, KEYS);
  k_topk<<<1, 256, 0, stream>>>(KEYS, 256, 256, MASK, nullptr, ORD);
  k_nms<<<1, 512, 0, stream>>>(RBOX, 5, 256, ORD, RVALID, 0.7f, 0);
  k_bbreg_rerec<<<1, 256, 0, stream>>>(RBOX, HREG, 256);
  // take_top O_ONET = 128
  k_keys<<<1, 256, 0, stream>>>(RBOX, 5, RVALID, 256, KEYS);
  k_topk<<<1, 256, 0, stream>>>(KEYS, 256, 128, MASK, VALS, IDXS);
  k_gather<<<1, 128, 0, stream>>>(RBOX, RVALID, IDXS, 128, OBOX, OVALID);

  // ================= Stage 3: O-Net on 128 x 48x48 crops =================
  k_crop<<<CDIV(128 * 48 * 48, 256), 256, 0, stream>>>(img, IMG_H, IMG_W,
                                                       OBOX, 128, 48, ACT_A, OKC);
  conv(ACT_A, 128, 48, 48, 16, Woc1, 3, 3, 160, 32, 32, o_c1.b, o_c1.a,
       ACT_B, nullptr, 46, 46);
  pool(ACT_B, 128, 46, 46, 32, ACT_C, 23, 23, 3, 2);
  conv(ACT_C, 128, 23, 23, 32, Woc2, 3, 3, 288, 64, 64, o_c2.b, o_c2.a,
       ACT_A, nullptr, 21, 21);
  pool(ACT_A, 128, 21, 21, 64, ACT_B, 10, 10, 3, 2);
  conv(ACT_B, 128, 10, 10, 64, Woc3, 3, 3, 576, 64, 64, o_c3.b, o_c3.a,
       ACT_C, nullptr, 8, 8);
  pool(ACT_C, 128, 8, 8, 64, ACT_A, 4, 4, 2, 2);
  conv(ACT_A, 128, 4, 4, 64, Woc4, 2, 2, 256, 128, 128, o_c4.b, o_c4.a,
       ACT_B, nullptr, 3, 3);
  k_flatten<<<CDIV(128 * 1152, 256), 256, 0, stream>>>(ACT_B, 128, 3, 3, 128, 128, ACT_C);
  conv(ACT_C, 128, 1, 1, 1152, Wod5, 1, 1, 1152, 256, 256, o_d5.b, o_d5.a,
       ACT_A, nullptr, 1, 1);
  conv(ACT_A, 128, 1, 1, 256, Wod61, 1, 1, 256, 16, 2, o_d61.b, nullptr,
       nullptr, HPROB, 1, 1);
  conv(ACT_A, 128, 1, 1, 256, Wod62, 1, 1, 256, 16, 4, o_d62.b, nullptr,
       nullptr, HREG, 1, 1);
  conv(ACT_A, 128, 1, 1, 256, Wod63, 1, 1, 256, 16, 10, o_d63.b, nullptr,
       nullptr, HPTS, 1, 1);
  k_onet_final<<<1, 128, 0, stream>>>(HPROB, HREG, HPTS, OKC, 0.7f,
                                      OBOX, OVALID, FBOX, FPTS);
  k_keys<<<1, 128, 0, stream>>>(FBOX, 5, OVALID, 128, KEYS);
  k_topk<<<1, 256, 0, stream>>>(KEYS, 128, 128, MASK, nullptr, ORD);
  k_nms<<<1, 512, 0, stream>>>(FBOX, 5, 128, ORD, OVALID, 0.7f, 1);  // mode=min
  k_output<<<1, 128, 0, stream>>>(FBOX, FPTS, OVALID, (float*)d_out);
}
